// TalkingHeadAttn_45930380263576
// MI455X (gfx1250) — matrix operand order repaired and checked by reading, not compiled, not run on hardware
//
#include <hip/hip_runtime.h>

// ---------------- problem constants ----------------
#define B_   8
#define N_   1024
#define C_   768
#define H_   16
#define HD_  48
#define HDP_ 64     // head dim padded to 64 for K=32 WMMA chains

typedef __attribute__((ext_vector_type(16))) _Float16 v16h;
typedef __attribute__((ext_vector_type(8)))  _Float16 v8h;
typedef __attribute__((ext_vector_type(8)))  float    v8f;
typedef __attribute__((ext_vector_type(4)))  unsigned v4u;
typedef __attribute__((ext_vector_type(8)))  int      v8i;
typedef __attribute__((ext_vector_type(4)))  int      v4i;

union V16U { v16h v; v8h h[2]; };
static __device__ inline v16h mk16(v8h lo, v8h hi){ V16U u; u.h[0]=lo; u.h[1]=hi; return u.v; }

static __device__ inline v8f wmma16(v16h a, v16h b, v8f c){
  // D = A(16x32 f16) * B(32x16 f16) + C(16x16 f32)
  return __builtin_amdgcn_wmma_f32_16x16x32_f16(false, a, false, b, (short)0, c, false, false);
}

// cross-lane reduction within each 16-lane half (wave32) via ds_swizzle xor
#define SWZ(x,m) __int_as_float(__builtin_amdgcn_ds_swizzle(__float_as_int(x), (((m)<<10)|0x1f)))
static __device__ inline float redmax16(float x){
  x = fmaxf(x, SWZ(x,1)); x = fmaxf(x, SWZ(x,2));
  x = fmaxf(x, SWZ(x,4)); x = fmaxf(x, SWZ(x,8));
  return x;
}
static __device__ inline float redsum16(float x){
  x += SWZ(x,1); x += SWZ(x,2); x += SWZ(x,4); x += SWZ(x,8);
  return x;
}

// ---------------- Tensor Data Mover (6-arg clang-23 builtin on this toolchain) ----------------
#if __has_builtin(__builtin_amdgcn_tensor_load_to_lds) && __has_builtin(__builtin_amdgcn_s_wait_tensorcnt)
#define HAVE_TDM 1
#else
#define HAVE_TDM 0
#endif

#if HAVE_TDM
// 2-D tile DMA: tile_d1 rows of tile_d0 elements (2-byte), row stride stride_d0 elems.
static __device__ inline void tdm_load_2d(void* lds_dst, const void* gsrc,
                                          unsigned tile_d0, unsigned tile_d1,
                                          unsigned long long tensor_d0, unsigned tensor_d1,
                                          unsigned long long stride_d0)
{
  unsigned long long ga = (unsigned long long)(size_t)gsrc;
  unsigned ldso = (unsigned)(size_t)lds_dst;         // low 32 bits = LDS byte offset
  v4u g0;
  g0.x = 1u;                                         // count=1, user descriptor
  g0.y = ldso;                                       // lds_addr (bytes)
  g0.z = (unsigned)ga;                               // global_addr[31:0]
  g0.w = (unsigned)((ga >> 32) & 0x1FFFFFFull) | (2u << 30);  // addr[56:32] | type=2
  // group1 bitfields (ISA 8.4): data_size=1 (2B), dims/strides in elements
  unsigned w0 = (1u << 16);
  unsigned w1 = (unsigned)((tensor_d0 & 0xFFFFull) << 16);                 // dim0[15:0] -> bits63:48
  unsigned w2 = (unsigned)((tensor_d0 >> 16) & 0xFFFFull)                  // dim0[31:16]-> bits79:64
              | ((tensor_d1 & 0xFFFFu) << 16);                             // dim1[15:0]
  unsigned w3 = ((tensor_d1 >> 16) & 0xFFFFu) | (tile_d0 << 16);           // dim1[31:16], tile0
  unsigned w4 = (tile_d1 & 0xFFFFu);                                       // tile1 (tile2=0)
  unsigned w5 = (unsigned)(stride_d0 & 0xFFFFFFFFull);                     // stride0[31:0]
  unsigned w6 = (unsigned)((stride_d0 >> 32) & 0xFFFFull);                 // stride0[47:32]
  v8i g1;
  g1[0]=(int)w0; g1[1]=(int)w1; g1[2]=(int)w2; g1[3]=(int)w3;
  g1[4]=(int)w4; g1[5]=(int)w5; g1[6]=(int)w6; g1[7]=0;
  v4i z4 = {0,0,0,0};
  v8i z8 = {0,0,0,0,0,0,0,0};
  __builtin_amdgcn_tensor_load_to_lds(g0, g1, z4, z4, z8, 0);
}
#endif

// ---------------- helper kernels ----------------
__global__ void k_cvt(const float* __restrict__ s, _Float16* __restrict__ d, int n){
  for (int i = blockIdx.x*blockDim.x + threadIdx.x; i < n; i += gridDim.x*blockDim.x)
    d[i] = (_Float16)s[i];
}

__global__ void k_zero(unsigned* __restrict__ p, int n){
  for (int i = blockIdx.x*blockDim.x + threadIdx.x; i < n; i += gridDim.x*blockDim.x)
    p[i] = 0u;
}

__global__ void k_prep_mix(const float* __restrict__ wl, const float* __restrict__ ww,
                           _Float16* __restrict__ Wlh, _Float16* __restrict__ Wwh){
  int t = threadIdx.x;                 // 512 threads
  int g = t >> 5, h = t & 31;
  Wlh[g*32+h] = (_Float16)(h < 16 ? wl[g*16+h] : 0.f);
  Wwh[g*32+h] = (_Float16)(h < 16 ? ww[g*16+h] : 0.f);
}

__global__ void k_vsum(const _Float16* __restrict__ VhT, float* __restrict__ vs){
  const int bh = blockIdx.x;
  const int t  = threadIdx.x;
  if (t < HD_){
    const _Float16* p = VhT + ((size_t)bh*HD_ + t)*N_;
    float s = 0.f;
    for (int k = 0; k < N_; ++k) s += (float)p[k];
    vs[bh*HD_ + t] = s;
  }
}

// ---------------- QKV GEMM: qkv = x @ w^T + b, scattered to Qh/Kh/VhT ----------------
__global__ __launch_bounds__(128) void k_qkv(
    const _Float16* __restrict__ X, const _Float16* __restrict__ W,
    const float* __restrict__ bias,
    _Float16* __restrict__ Qh, _Float16* __restrict__ Kh, _Float16* __restrict__ VhT)
{
  __shared__ __align__(16) _Float16 Alds[16][C_];   // 24KB, shared A tile for 4 waves

  const int tid = threadIdx.x, lane = tid & 31, wv = tid >> 5;
  const int lm = lane & 15, lg = lane >> 4;
  const int m0 = blockIdx.y * 16;
  const int o0 = blockIdx.x * 256 + wv * 64;

#if HAVE_TDM
  if (wv == 0){
    tdm_load_2d(&Alds[0][0], X + (size_t)m0*C_, 16u*C_, 1u,
                (unsigned long long)16*C_, 1u, (unsigned long long)16*C_);
    __builtin_amdgcn_s_wait_tensorcnt(0);
  }
#else
  for (int i = tid; i < 16*C_/2; i += 128)
    ((unsigned*)Alds)[i] = ((const unsigned*)(X + (size_t)m0*C_))[i];
#endif
  __syncthreads();

  v8f acc[4];
  { v8f z = {};
    #pragma unroll
    for (int i = 0; i < 4; ++i) acc[i] = z; }

  for (int k0 = 0; k0 < C_; k0 += 32){
    const _Float16* xa = &Alds[lm][k0 + lg*16];
    v16h A = mk16(*(const v8h*)xa, *(const v8h*)(xa + 8));
    #pragma unroll
    for (int nb = 0; nb < 4; ++nb){
      const _Float16* wb = W + (size_t)(o0 + nb*16 + lm)*C_ + k0 + lg*16;
      v16h Bv = mk16(*(const v8h*)wb, *(const v8h*)(wb + 8));
      acc[nb] = wmma16(A, Bv, acc[nb]);
    }
  }

  const float qscale = 0.14433756729740643f;   // 48^-0.5
  #pragma unroll
  for (int nb = 0; nb < 4; ++nb){
    const int o = o0 + nb*16 + lm;
    const float bv = bias[o];
    #pragma unroll
    for (int r = 0; r < 8; ++r){
      const int n  = m0 + r + 8*lg;
      const int bI = n >> 10, nr = n & 1023;
      float val = acc[nb][r] + bv;
      if (o < C_){
        int h = o / HD_, d = o % HD_;
        Qh[(((size_t)(bI*H_ + h))*N_ + nr)*HDP_ + d] = (_Float16)(val * qscale);
      } else if (o < 2*C_){
        int oo = o - C_; int h = oo / HD_, d = oo % HD_;
        Kh[(((size_t)(bI*H_ + h))*N_ + nr)*HDP_ + d] = (_Float16)val;
      } else {
        int oo = o - 2*C_; int h = oo / HD_, d = oo % HD_;
        VhT[(((size_t)(bI*H_ + h))*HD_ + d)*N_ + nr] = (_Float16)val;  // transposed store
      }
    }
  }
}

// ---------------- fused talking-heads attention ----------------
// 1 workgroup (4 waves) per (batch, 16-row q tile), all 16 heads.
// Dynamic LDS: Q tiles (TDM-staged, 32KB) + S0/P1/P2 staging + softmax stats = 83KB.
__global__ __launch_bounds__(128) void k_attn(
    const _Float16* __restrict__ Qh, const _Float16* __restrict__ Kh,
    const _Float16* __restrict__ VhT, const float* __restrict__ vsum,
    const _Float16* __restrict__ Wlh, const _Float16* __restrict__ Wwh,
    const float* __restrict__ bl, const float* __restrict__ bw,
    _Float16* __restrict__ AttnOut)
{
  extern __shared__ __align__(16) char smem[];
  _Float16 (*Qlds)[16][HDP_] = (_Float16(*)[16][HDP_])(smem);           // [16][16][64] 32KB
  _Float16 (*S0)[16][32]     = (_Float16(*)[16][32])(smem + 32768);     // raw scores [q][k][h]
  _Float16 (*P1)[16][32]     = (_Float16(*)[16][32])(smem + 49152);     // softmaxed  [q][k][g]
  _Float16 (*P2)[16][32]     = (_Float16(*)[16][32])(smem + 65536);     // mixed      [g'][q][k32]
  float* mrow = (float*)(smem + 81920);   // [g*16+q]
  float* lrow = mrow + 256;
  float* linv = lrow + 256;

  const int tid = threadIdx.x;
  const int lane = tid & 31;
  const int wv = tid >> 5;
  const int lm = lane & 15;
  const int lg = lane >> 4;
  const int b  = blockIdx.x >> 6;
  const int q0 = (blockIdx.x & 63) << 4;

  // stage all 16 Q head-tiles (each 16x64 f16 = 1024 contiguous elems) via one TDM descriptor
#if HAVE_TDM
  if (wv == 0){
    tdm_load_2d(&Qlds[0][0][0], Qh + ((size_t)(b*H_)*N_ + q0)*HDP_,
                16u*HDP_ /*tile_d0 = 1024*/, 16u /*heads*/,
                (unsigned long long)N_*HDP_, 16u, (unsigned long long)N_*HDP_);
    __builtin_amdgcn_s_wait_tensorcnt(0);
  }
#else
  for (int i = tid; i < 16*512; i += 128){           // u32 units: 16 heads x 512 dwords
    int h = i >> 9, j = i & 511;
    ((unsigned*)Qlds)[i] =
      ((const unsigned*)(Qh + (((size_t)(b*H_ + h))*N_ + q0)*HDP_))[j];
  }
#endif

  for (int i = tid; i < 256; i += 128){ mrow[i] = -1e30f; lrow[i] = 0.f; }
  for (int i = tid; i < 16*16*16; i += 128){   // zero padded K-halves once
    int q = i >> 8, k = (i >> 4) & 15, h = i & 15;
    S0[q][k][16+h] = (_Float16)0.f;
    P1[q][k][16+h] = (_Float16)0.f;
  }
  __syncthreads();

  // resident WMMA A-operands: padded w_l / w_w, plus b_l as C operand
  v16h Awl, Aww; v8f cbl;
  {
    const _Float16* pl = Wlh + lm*32 + lg*16;
    Awl = mk16(*(const v8h*)pl, *(const v8h*)(pl + 8));
    const _Float16* pw = Wwh + lm*32 + lg*16;
    Aww = mk16(*(const v8h*)pw, *(const v8h*)(pw + 8));
    #pragma unroll
    for (int r = 0; r < 8; ++r) cbl[r] = bl[r + 8*lg];
  }

  // raw scores for this wave's 4 heads for one 16-wide k tile -> S0
  auto s_stage = [&](int k0){
    #pragma unroll
    for (int hh = 0; hh < 4; ++hh){
      const int h = wv*4 + hh;
      const _Float16* qb = &Qlds[h][lm][0];
      v16h A0 = mk16(*(const v8h*)(qb + lg*16),      *(const v8h*)(qb + lg*16 + 8));
      v16h A1 = mk16(*(const v8h*)(qb + 32 + lg*16), *(const v8h*)(qb + 32 + lg*16 + 8));
      const _Float16* kb = Kh + (((size_t)(b*H_ + h))*N_ + (k0 + lm))*HDP_;
      v16h B0 = mk16(*(const v8h*)(kb + lg*16),      *(const v8h*)(kb + lg*16 + 8));
      v16h B1 = mk16(*(const v8h*)(kb + 32 + lg*16), *(const v8h*)(kb + 32 + lg*16 + 8));
      v8f s = {};
      s = wmma16(A0, B0, s);
      s = wmma16(A1, B1, s);
      #pragma unroll
      for (int r = 0; r < 8; ++r)
        S0[r + 8*lg][lm][h] = (_Float16)s[r];
    }
  };

  // head-mix #1 as WMMA: S1[g,k] (at q=cb) = w_l @ S0 + b_l
  auto mix1 = [&](int cb)->v8f{
    const _Float16* sb = &S0[cb][lm][lg*16];
    v16h Bm = mk16(*(const v8h*)sb, *(const v8h*)(sb + 8));
    return wmma16(Awl, Bm, cbl);
  };

  // ---- pass A: online softmax statistics ----
  for (int k0 = 0; k0 < N_; k0 += 16){
    s_stage(k0);
    __syncthreads();
    #pragma unroll
    for (int i = 0; i < 4; ++i){
      const int cb = wv*4 + i;
      v8f s1 = mix1(cb);
      float t[8], es[8];
      #pragma unroll
      for (int r = 0; r < 8; ++r) t[r] = redmax16(s1[r]);
      #pragma unroll
      for (int r = 0; r < 8; ++r) es[r] = redsum16(__expf(s1[r] - t[r]));
      if (lm == 0){
        #pragma unroll
        for (int r = 0; r < 8; ++r){
          const int g = r + 8*lg;
          float mo = mrow[g*16 + cb];
          float mn = fmaxf(mo, t[r]);
          lrow[g*16 + cb] = lrow[g*16 + cb]*__expf(mo - mn) + es[r]*__expf(t[r] - mn);
          mrow[g*16 + cb] = mn;
        }
      }
    }
    __syncthreads();
  }
  for (int i = tid; i < 256; i += 128) linv[i] = 1.f / lrow[i];
  __syncthreads();

  // ---- pass B: recompute, normalize, mix #2, accumulate P@V ----
  v8f acc[4][3];
  { v8f z = {};
    #pragma unroll
    for (int a = 0; a < 4; ++a)
      #pragma unroll
      for (int d = 0; d < 3; ++d) acc[a][d] = z; }

  for (int k0 = 0; k0 < N_; k0 += 32){
    #pragma unroll
    for (int ks = 0; ks < 32; ks += 16){
      s_stage(k0 + ks);
      __syncthreads();
      #pragma unroll
      for (int i = 0; i < 4; ++i){
        const int cb = wv*4 + i;
        v8f s1 = mix1(cb);
        #pragma unroll
        for (int r = 0; r < 8; ++r){
          const int g = r + 8*lg;
          float pval = __expf(s1[r] - mrow[g*16 + cb]) * linv[g*16 + cb];
          P1[cb][lm][g] = (_Float16)pval;
        }
      }
      __syncthreads();
      #pragma unroll
      for (int i = 0; i < 4; ++i){
        const int cb = wv*4 + i;
        const _Float16* pb = &P1[cb][lm][lg*16];
        v16h Bp = mk16(*(const v8h*)pb, *(const v8h*)(pb + 8));
        v8f z = {};
        v8f p2 = wmma16(Aww, Bp, z);
        #pragma unroll
        for (int r = 0; r < 8; ++r)
          P2[r + 8*lg][cb][ks + lm] = (_Float16)p2[r];
      }
      __syncthreads();
    }
    // V stage: out[g',q,d] += P2[g',q,k32] @ V[g',k32,d]
    #pragma unroll
    for (int hh = 0; hh < 4; ++hh){
      const int h = wv*4 + hh;
      const _Float16* pa = &P2[h][lm][lg*16];
      v16h Ap = mk16(*(const v8h*)pa, *(const v8h*)(pa + 8));
      #pragma unroll
      for (int db = 0; db < 3; ++db){
        const _Float16* vb = VhT + (((size_t)(b*H_ + h))*HD_ + db*16 + lm)*N_ + k0 + lg*16;
        v16h Bv = mk16(*(const v8h*)vb, *(const v8h*)(vb + 8));
        acc[hh][db] = wmma16(Ap, Bv, acc[hh][db]);
      }
    }
    __syncthreads();
  }

  // epilogue: + b_w * sum_k v, write f16 for the proj GEMM
  #pragma unroll
  for (int hh = 0; hh < 4; ++hh){
    const int h = wv*4 + hh;
    const float bwv = bw[h];
    #pragma unroll
    for (int db = 0; db < 3; ++db){
      const int d = db*16 + lm;
      const float vs = vsum[(b*H_ + h)*HD_ + d];
      #pragma unroll
      for (int r = 0; r < 8; ++r){
        const int q = q0 + r + 8*lg;
        AttnOut[((size_t)b*N_ + q)*C_ + h*HD_ + d] = (_Float16)(acc[hh][db][r] + bwv*vs);
      }
    }
  }
}

// ---------------- output projection GEMM (f32 out) ----------------
__global__ __launch_bounds__(128) void k_proj(
    const _Float16* __restrict__ Ain, const _Float16* __restrict__ W,
    const float* __restrict__ bias, float* __restrict__ Out)
{
  __shared__ __align__(16) _Float16 Alds[16][C_];

  const int tid = threadIdx.x, lane = tid & 31, wv = tid >> 5;
  const int lm = lane & 15, lg = lane >> 4;
  const int m0 = blockIdx.y * 16;
  const int o0 = blockIdx.x * 256 + wv * 64;

#if HAVE_TDM
  if (wv == 0){
    tdm_load_2d(&Alds[0][0], Ain + (size_t)m0*C_, 16u*C_, 1u,
                (unsigned long long)16*C_, 1u, (unsigned long long)16*C_);
    __builtin_amdgcn_s_wait_tensorcnt(0);
  }
#else
  for (int i = tid; i < 16*C_/2; i += 128)
    ((unsigned*)Alds)[i] = ((const unsigned*)(Ain + (size_t)m0*C_))[i];
#endif
  __syncthreads();

  v8f acc[4];
  { v8f z = {};
    #pragma unroll
    for (int i = 0; i < 4; ++i) acc[i] = z; }

  for (int k0 = 0; k0 < C_; k0 += 32){
    const _Float16* xa = &Alds[lm][k0 + lg*16];
    v16h A = mk16(*(const v8h*)xa, *(const v8h*)(xa + 8));
    #pragma unroll
    for (int nb = 0; nb < 4; ++nb){
      const _Float16* wb = W + (size_t)(o0 + nb*16 + lm)*C_ + k0 + lg*16;
      v16h Bv = mk16(*(const v8h*)wb, *(const v8h*)(wb + 8));
      acc[nb] = wmma16(A, Bv, acc[nb]);
    }
  }
  #pragma unroll
  for (int nb = 0; nb < 4; ++nb){
    const int o = o0 + nb*16 + lm;
    const float bv = bias[o];
    #pragma unroll
    for (int r = 0; r < 8; ++r){
      const int n = m0 + r + 8*lg;
      Out[(size_t)n*C_ + o] = acc[nb][r] + bv;
    }
  }
}

// ---------------- launcher ----------------
extern "C" void kernel_launch(void* const* d_in, const int* in_sizes, int n_in,
                              void* d_out, int out_size, void* d_ws, size_t ws_size,
                              hipStream_t stream)
{
  const float* x      = (const float*)d_in[0];
  const float* w_qkv  = (const float*)d_in[1];
  const float* b_qkv  = (const float*)d_in[2];
  const float* w_l    = (const float*)d_in[3];
  const float* b_l    = (const float*)d_in[4];
  const float* w_w    = (const float*)d_in[5];
  const float* b_w    = (const float*)d_in[6];
  const float* w_proj = (const float*)d_in[7];
  const float* b_proj = (const float*)d_in[8];

  char* p = (char*)d_ws;
  auto carve = [&](size_t bytes)->char*{
    char* r = p; p += (bytes + 255) & ~(size_t)255; return r; };

  _Float16* xh     = (_Float16*)carve((size_t)B_*N_*C_*2);
  _Float16* wqkvh  = (_Float16*)carve((size_t)3*C_*C_*2);
  _Float16* wprojh = (_Float16*)carve((size_t)C_*C_*2);
  _Float16* Qh     = (_Float16*)carve((size_t)B_*H_*N_*HDP_*2);
  _Float16* Kh     = (_Float16*)carve((size_t)B_*H_*N_*HDP_*2);
  _Float16* VhT    = (_Float16*)carve((size_t)B_*H_*HD_*N_*2);
  _Float16* attn_o = (_Float16*)carve((size_t)B_*N_*C_*2);
  _Float16* Wlh    = (_Float16*)carve(16*32*2);
  _Float16* Wwh    = (_Float16*)carve(16*32*2);
  float*    vsum   = (float*)carve((size_t)B_*H_*HD_*4);

  k_cvt<<<2048, 256, 0, stream>>>(x, xh, B_*N_*C_);
  k_cvt<<<2048, 256, 0, stream>>>(w_qkv, wqkvh, 3*C_*C_);
  k_cvt<<<1024, 256, 0, stream>>>(w_proj, wprojh, C_*C_);
  k_prep_mix<<<1, 512, 0, stream>>>(w_l, w_w, Wlh, Wwh);
  k_zero<<<2048, 256, 0, stream>>>((unsigned*)Qh, B_*H_*N_*HDP_/2);
  k_zero<<<2048, 256, 0, stream>>>((unsigned*)Kh, B_*H_*N_*HDP_/2);

  k_qkv<<<dim3(9, 512), 128, 0, stream>>>(xh, wqkvh, b_qkv, Qh, Kh, VhT);
  k_vsum<<<B_*H_, 64, 0, stream>>>(VhT, vsum);
  k_attn<<<B_*(N_/16), 128, 84992, stream>>>(Qh, Kh, VhT, vsum, Wlh, Wwh, b_l, b_w, attn_o);
  k_proj<<<dim3(3, 512), 128, 0, stream>>>(attn_o, wprojh, b_proj, (float*)d_out);
}
